// GraphConvNetworkLayer_1211180778248
// MI455X (gfx1250) — matrix operand order repaired and checked by reading, not compiled, not run on hardware
//
#include <hip/hip_runtime.h>
#include <hip/hip_bf16.h>
#include <math.h>

typedef __attribute__((ext_vector_type(16))) _Float16 v16h;
typedef __attribute__((ext_vector_type(8)))  _Float16 v8h;
typedef __attribute__((ext_vector_type(8)))  float    v8f;
typedef __attribute__((ext_vector_type(4)))  float    v4f;
typedef __attribute__((ext_vector_type(4)))  int      v4i;

#define NN 8192
#define FF 256
#define NSLOPE 0.2f
#define SPLIT 4
#define JCHUNK (NN / SPLIT)   // 2048 columns per split block

// Pointer flavors for the async-to-LDS builtin (param type is v4i* per the
// compiler diagnostic): global = addrspace(1), LDS = addrspace(3).
typedef __attribute__((address_space(1))) v4i glb_v4i;
typedef __attribute__((address_space(3))) v4i lds_v4i;

// ---------------------------------------------------------------------------
// Prep: transpose W (f32 [k][n]) -> WT16 (f16 [n][k]); zero wh1/wh2.
// ---------------------------------------------------------------------------
__global__ void gat_prep(const float* __restrict__ W, _Float16* __restrict__ WT16,
                         float* __restrict__ wh1, float* __restrict__ wh2) {
  int idx = blockIdx.x * 256 + threadIdx.x;   // 65536 threads
  int n = idx >> 8, k = idx & 255;
  WT16[n * FF + k] = (_Float16)W[k * FF + n];
  if (idx < NN) { wh1[idx] = 0.f; wh2[idx] = 0.f; }
}

// Zero the split-accumulation buffers (harness poisons ws with 0xAA).
__global__ void gat_zero(float* __restrict__ accG, float* __restrict__ lsumG) {
  size_t idx = (size_t)blockIdx.x * 256 + threadIdx.x;   // NN*FF threads
  accG[idx] = 0.f;
  if (idx < NN) lsumG[idx] = 0.f;
}

// ---------------------------------------------------------------------------
// GEMM1: Wh = h @ W via v_wmma_f32_16x16x32_f16, one 16x16 tile per wave.
// Emits WhT16 (transposed f16 Wh = B-operand layout for the attention GEMM)
// and wh1/wh2 = Wh@a1 / Wh@a2 via f32 atomics.
// ---------------------------------------------------------------------------
__global__ void __launch_bounds__(256)
gat_gemm1(const float* __restrict__ h, const _Float16* __restrict__ WT16,
          const float* __restrict__ avec, _Float16* __restrict__ WhT16,
          float* __restrict__ wh1, float* __restrict__ wh2) {
  const int lane = threadIdx.x & 31;
  const int wave = threadIdx.x >> 5;
  const int tile = blockIdx.x * 8 + wave;        // 512 M-tiles x 16 N-tiles
  const int m0   = (tile >> 4) << 4;
  const int n0   = (tile & 15) << 4;
  const int row  = lane & 15;
  const int half = lane >> 4;

  const float*    hr = h    + (size_t)(m0 + row) * FF;
  const _Float16* wt = WT16 + (size_t)(n0 + row) * FF;

  v8f c = {0.f, 0.f, 0.f, 0.f, 0.f, 0.f, 0.f, 0.f};

#pragma unroll
  for (int k0 = 0; k0 < FF; k0 += 32) {
    v4f a0 = *(const v4f*)(hr + k0 + 16 * half);
    v4f a1 = *(const v4f*)(hr + k0 + 16 * half + 4);
    v4f a2 = *(const v4f*)(hr + k0 + 16 * half + 8);
    v4f a3 = *(const v4f*)(hr + k0 + 16 * half + 12);
    v16h A;
#pragma unroll
    for (int e = 0; e < 4; ++e) {
      A[e]      = (_Float16)a0[e];
      A[e + 4]  = (_Float16)a1[e];
      A[e + 8]  = (_Float16)a2[e];
      A[e + 12] = (_Float16)a3[e];
    }
    v16h B = *(const v16h*)(wt + k0 + 16 * half);
    c = __builtin_amdgcn_wmma_f32_16x16x32_f16(false, A, false, B, (short)0, c,
                                               false, false);
  }

  {
    v8h o;
#pragma unroll
    for (int r = 0; r < 8; ++r) o[r] = (_Float16)c[r];
    *(v8h*)(WhT16 + (size_t)(n0 + row) * NN + m0 + 8 * half) = o;
  }

  const float a1c = avec[n0 + row];
  const float a2c = avec[FF + n0 + row];
#pragma unroll
  for (int r = 0; r < 8; ++r) {
    int m = m0 + r + 8 * half;
    atomicAdd(&wh1[m], c[r] * a1c);
    atomicAdd(&wh2[m], c[r] * a2c);
  }
}

// ---------------------------------------------------------------------------
// Global max of wh2 (softmax stabilizer upper bound, lrelu is monotone).
// ---------------------------------------------------------------------------
__global__ void gat_max(const float* __restrict__ wh2, float* __restrict__ mx) {
  __shared__ float red[1024];
  float m = -3.4e38f;
  for (int i = threadIdx.x; i < NN; i += 1024) m = fmaxf(m, wh2[i]);
  red[threadIdx.x] = m;
  __syncthreads();
  for (int s = 512; s > 0; s >>= 1) {
    if ((int)threadIdx.x < s) red[threadIdx.x] = fmaxf(red[threadIdx.x], red[threadIdx.x + s]);
    __syncthreads();
  }
  if (threadIdx.x == 0) mx[0] = red[0];
}

// ---------------------------------------------------------------------------
// Attention, j-split 4x: each block handles 64 rows x 2048 cols.
// Per wave: 16 rows, P(16x32) built on the fly -> 16 WMMA per step.
// Accumulates unnormalized acc / l partials with f32 atomics; fuses the adj
// passthrough so adj is read exactly once from HBM.
// ---------------------------------------------------------------------------
__global__ void __launch_bounds__(128)
gat_attn(const int* __restrict__ adj, const _Float16* __restrict__ WhT16,
         const float* __restrict__ wh1, const float* __restrict__ wh2,
         const float* __restrict__ mx, float* __restrict__ accG,
         float* __restrict__ lsumG, float* __restrict__ adj_out) {
  __shared__ float swh2[JCHUNK];  // 8 KB slice of wh2
  const int split = blockIdx.x & (SPLIT - 1);
  const int mblk  = blockIdx.x >> 2;
  const int jbase = split * JCHUNK;

  // Stage wh2 slice into LDS via the CDNA5 async-to-LDS path when available.
#if __has_builtin(__builtin_amdgcn_global_load_async_to_lds_b128)
  for (int t = threadIdx.x * 4; t < JCHUNK; t += 128 * 4) {
    glb_v4i* gp = (glb_v4i*)(uintptr_t)(wh2 + jbase + t);
    // low 32 bits of a flat shared address == LDS byte offset
    lds_v4i* lp = (lds_v4i*)(uint32_t)(uintptr_t)(&swh2[t]);
    __builtin_amdgcn_global_load_async_to_lds_b128(gp, lp, 0, 0);
  }
#if __has_builtin(__builtin_amdgcn_s_wait_asynccnt)
  __builtin_amdgcn_s_wait_asynccnt(0);
#else
  asm volatile("s_wait_asynccnt 0" ::: "memory");
#endif
#else
  for (int t = threadIdx.x; t < JCHUNK; t += 128) swh2[t] = wh2[jbase + t];
#endif
  __syncthreads();

  const int lane = threadIdx.x & 31;
  const int wave = threadIdx.x >> 5;
  const int m0   = mblk * 64 + wave * 16;
  const int row  = lane & 15;
  const int half = lane >> 4;
  const int i    = m0 + row;

  const float M2 = mx[0];
  const float w1 = wh1[i];
  const float mt = w1 + M2;
  const float mi = mt > 0.f ? mt : NSLOPE * mt;  // >= every logit of row i

  const int* arow = adj + (size_t)i * NN + jbase;
  float*     orow = adj_out + (size_t)i * NN + jbase;

  v8f acc[16];
#pragma unroll
  for (int t = 0; t < 16; ++t)
    acc[t] = (v8f){0.f, 0.f, 0.f, 0.f, 0.f, 0.f, 0.f, 0.f};

  float lsum = 0.f;

  for (int jl = 0; jl < JCHUNK; jl += 32) {
    const int o1 = jl + 16 * half;
    const int o2 = jl + 16 * half + 8;
    v4i ad0 = *(const v4i*)(arow + o1);
    v4i ad1 = *(const v4i*)(arow + o1 + 4);
    v4i ad2 = *(const v4i*)(arow + o2);
    v4i ad3 = *(const v4i*)(arow + o2 + 4);
    __builtin_prefetch(arow + jl + 256, 0, 1);  // global_prefetch_b8 read-ahead

    int ads[16];
#pragma unroll
    for (int e = 0; e < 4; ++e) {
      ads[e] = ad0[e]; ads[e + 4] = ad1[e]; ads[e + 8] = ad2[e]; ads[e + 12] = ad3[e];
    }

    v16h P;
#pragma unroll
    for (int e = 0; e < 16; ++e) {
      int jloc = (e < 8) ? (o1 + e) : (o2 + e - 8);
      float s  = w1 + swh2[jloc];
      s        = s > 0.f ? s : NSLOPE * s;
      float p  = (ads[e] > 0) ? __expf(s - mi) : 0.f;  // exp <= 1 by construction
      lsum += p;
      P[e] = (_Float16)p;
    }

    // Fused adj passthrough (second tuple output), as float
    v4f f0, f1, f2, f3;
#pragma unroll
    for (int e = 0; e < 4; ++e) {
      f0[e] = (float)ad0[e]; f1[e] = (float)ad1[e];
      f2[e] = (float)ad2[e]; f3[e] = (float)ad3[e];
    }
    *(v4f*)(orow + o1)     = f0;
    *(v4f*)(orow + o1 + 4) = f1;
    *(v4f*)(orow + o2)     = f2;
    *(v4f*)(orow + o2 + 4) = f3;

    // acc[t] += P(16x32) @ WhT tile(32x16), t over 256 output features
#pragma unroll
    for (int t = 0; t < 16; ++t) {
      const _Float16* bp =
          WhT16 + (size_t)(t * 16 + row) * NN + jbase + jl + 16 * half;
      v16h B = *(const v16h*)bp;
      acc[t] = __builtin_amdgcn_wmma_f32_16x16x32_f16(false, P, false, B, (short)0,
                                                      acc[t], false, false);
    }
  }

  // Row partial denominator: combine the two K-halves (lane ^ 16), add once.
  const float lf = lsum + __shfl_xor(lsum, 16, 32);
  if (lane < 16) atomicAdd(&lsumG[i], lf);

#pragma unroll
  for (int r = 0; r < 8; ++r) {
#pragma unroll
    for (int t = 0; t < 16; ++t) {
      atomicAdd(&accG[(size_t)(m0 + r + 8 * half) * FF + t * 16 + row], acc[t][r]);
    }
  }
}

// ---------------------------------------------------------------------------
// Finalize: out = elu(acc / l)
// ---------------------------------------------------------------------------
__global__ void gat_final(const float* __restrict__ accG,
                          const float* __restrict__ lsumG,
                          float* __restrict__ out) {
  size_t idx = (size_t)blockIdx.x * 256 + threadIdx.x;  // NN*FF threads
  int i = (int)(idx >> 8);
  float l   = lsumG[i];
  float inv = l > 0.f ? 1.f / l : 0.f;
  float hp  = accG[idx] * inv;
  out[idx]  = hp > 0.f ? hp : (__expf(hp) - 1.f);
}

// ---------------------------------------------------------------------------
extern "C" void kernel_launch(void* const* d_in, const int* in_sizes, int n_in,
                              void* d_out, int out_size, void* d_ws, size_t ws_size,
                              hipStream_t stream) {
  (void)in_sizes; (void)n_in; (void)out_size; (void)ws_size;
  const float* h    = (const float*)d_in[0];
  const int*   adj  = (const int*)d_in[1];
  const float* W    = (const float*)d_in[2];
  const float* avec = (const float*)d_in[3];
  float* out = (float*)d_out;

  // Workspace layout (~12.3 MiB)
  char* ws = (char*)d_ws;
  _Float16* WhT16 = (_Float16*)ws;                          // 4 MiB
  float* wh1  = (float*)(ws + (size_t)FF * NN * sizeof(_Float16));
  float* wh2  = wh1 + NN;
  float* mx   = wh2 + NN;
  _Float16* WT16 = (_Float16*)(mx + 16);                    // 128 KiB
  float* accG  = (float*)((char*)WT16 + (size_t)FF * FF * sizeof(_Float16));  // 8 MiB
  float* lsumG = accG + (size_t)NN * FF;

  gat_prep <<<256, 256, 0, stream>>>(W, WT16, wh1, wh2);
  gat_zero <<<(NN * FF) / 256, 256, 0, stream>>>(accG, lsumG);
  gat_gemm1<<<1024, 256, 0, stream>>>(h, WT16, avec, WhT16, wh1, wh2);
  gat_max  <<<1, 1024, 0, stream>>>(wh2, mx);
  gat_attn <<<128 * SPLIT, 128, 0, stream>>>(adj, WhT16, wh1, wh2, mx,
                                             accG, lsumG,
                                             out + (size_t)NN * FF);
  gat_final<<<(NN * FF) / 256, 256, 0, stream>>>(accG, lsumG, out);
}
